// SASRecMoESoft_69930657513758
// MI455X (gfx1250) — matrix operand order repaired and checked by reading, not compiled, not run on hardware
//
#include <hip/hip_runtime.h>
#include <hip/hip_bf16.h>
#include <math.h>
#include <stdint.h>

// ---------------- problem constants ----------------
#define B_   128
#define L_   200
#define D_   50
#define H_   5
#define E_   10
#define NB_  2
#define DH_  10
#define BL_  (B_*L_)      // 25600 rows, divisible by 16
#define DP_  64           // D padded to 2 WMMA k-steps
#define LP_  208          // L padded to 13 16-tiles
#define SQRTD     7.0710678118654755f
#define RSQRT_DH  0.31622776601683794f

typedef __bf16 bf16_t;
typedef bf16_t v16bf __attribute__((ext_vector_type(16)));
typedef float  v8f   __attribute__((ext_vector_type(8)));

static __device__ __forceinline__ v8f wmma_bf16(v16bf a, v16bf b, v8f c) {
  // D = A(16x32 bf16) x B(32x16 bf16) + C(16x16 f32)
  return __builtin_amdgcn_wmma_f32_16x16x32_bf16(false, a, false, b, (short)0, c, false, false);
}

// ---------------- CDNA5 async global->LDS path ----------------
// GLOBAL_LOAD_ASYNC_TO_LDS_B32: memory -> LDS with no VGPR staging (ASYNCcnt).
typedef __attribute__((address_space(1))) int glb_int;
typedef __attribute__((address_space(3))) int lds_int;

#if __has_builtin(__builtin_amdgcn_global_load_async_to_lds_b32)
#define HAVE_ASYNC_LDS 1
#else
#define HAVE_ASYNC_LDS 0
#endif

static __device__ __forceinline__ void async_g2l_b32(const void* g, void* l) {
#if HAVE_ASYNC_LDS
  __builtin_amdgcn_global_load_async_to_lds_b32((glb_int*)g, (lds_int*)l, 0, 0);
#else
  *(uint32_t*)l = *(const uint32_t*)g;   // fallback: staged copy
#endif
}
static __device__ __forceinline__ void async_wait_all() {
#if HAVE_ASYNC_LDS
  asm volatile("s_wait_asynccnt 0" ::: "memory");
#endif
}

// A fragment, 16x32 bf16 tile at base (row-major, leading dim ld).
// Contiguous k order used here: lanes 0-15 row M=lane, K 0..15; lanes 16-31 same M, K 16..31.
static __device__ __forceinline__ v16bf load_a_frag(const bf16_t* base, int ld) {
  int l  = threadIdx.x & 31;
  int m  = l & 15;
  int hi = (l >> 4) * 16;
  v16bf a;
#pragma unroll
  for (int p = 0; p < 8; ++p) {
    int k = hi + 2 * p;
    a[2 * p]     = base[m * ld + k];
    a[2 * p + 1] = base[m * ld + k + 1];
  }
  return a;
}

// B fragment, 32x16 bf16 tile at base (row-major K x N, leading dim ld).
// Contiguous k order used here: lane holds column n = lane&15; halves j -> K = j + (lane>=16)*16.
static __device__ __forceinline__ v16bf load_b_frag_rows(const bf16_t* base, int ld) {
  int l  = threadIdx.x & 31;
  int n  = l & 15;
  int kb = (l >> 4) * 16;
  v16bf b;
#pragma unroll
  for (int j = 0; j < 16; ++j) b[j] = base[(kb + j) * ld + n];
  return b;
}

// B fragment from pre-packed weights: 512 bf16 = [lane][16 halves], contiguous per lane.
static __device__ __forceinline__ v16bf load_b_frag_packed(const bf16_t* frag) {
  int l = threadIdx.x & 31;
  const bf16_t* p = frag + l * 16;
  v16bf b;
#pragma unroll
  for (int j = 0; j < 16; ++j) b[j] = p[j];
  return b;
}

static __device__ __forceinline__ float wred_sum(float v) {
#pragma unroll
  for (int o = 16; o > 0; o >>= 1) v += __shfl_xor(v, o, 32);
  return v;
}
static __device__ __forceinline__ float wred_max(float v) {
#pragma unroll
  for (int o = 16; o > 0; o >>= 1) v = fmaxf(v, __shfl_xor(v, o, 32));
  return v;
}

// ---------------- stage kernels ----------------

// seqs[b,l,:] = item_emb[log]*sqrt(D) + pos_emb[(l+1)*(log!=0)]
__global__ __launch_bounds__(64)
void k_embed(const float* __restrict__ item_emb, const float* __restrict__ pos_emb,
             const int* __restrict__ log_seqs, float* __restrict__ seqs) {
  int lpos = blockIdx.x, b = blockIdx.y;
  int d = threadIdx.x;
  if (d >= D_) return;
  int it = log_seqs[b * L_ + lpos];
  int pp = (it != 0) ? (lpos + 1) : 0;
  seqs[(b * L_ + lpos) * D_ + d] = item_emb[it * D_ + d] * SQRTD + pos_emb[pp * D_ + d];
}

// pooled[b,d] = mean_l item_emb[log]*sqrt(D)  (pre-positional, per reference)
__global__ __launch_bounds__(64)
void k_pool(const float* __restrict__ item_emb, const int* __restrict__ log_seqs,
            float* __restrict__ pooled) {
  int b = blockIdx.x, d = threadIdx.x;
  if (d >= D_) return;
  float s = 0.f;
  for (int l = 0; l < L_; ++l) s += item_emb[log_seqs[b * L_ + l] * D_ + d];
  pooled[b * D_ + d] = s * SQRTD * (1.0f / L_);
}

// gate[b,:] = softmax(relu(pooled@rW1^T+rb1)@rW2^T+rb2)
__global__ __launch_bounds__(128)
void k_router(const float* __restrict__ pooled, const float* __restrict__ rW1,
              const float* __restrict__ rb1, const float* __restrict__ rW2,
              const float* __restrict__ rb2, float* __restrict__ gate) {
  __shared__ float h[100];
  __shared__ float lg[E_];
  int b = blockIdx.x, t = threadIdx.x;
  if (t < 100) {
    float a = rb1[t];
    for (int d = 0; d < D_; ++d) a += pooled[b * D_ + d] * rW1[t * D_ + d];
    h[t] = fmaxf(a, 0.f);
  }
  __syncthreads();
  if (t < E_) {
    float a = rb2[t];
    for (int j = 0; j < 100; ++j) a += rW2[t * 100 + j] * h[j];
    lg[t] = a;
  }
  __syncthreads();
  if (t == 0) {
    float mx = -1e30f;
    for (int e = 0; e < E_; ++e) mx = fmaxf(mx, lg[e]);
    float s = 0.f;
    for (int e = 0; e < E_; ++e) { lg[e] = __expf(lg[e] - mx); s += lg[e]; }
    float inv = 1.0f / s;
    for (int e = 0; e < E_; ++e) gate[b * E_ + e] = lg[e] * inv;
  }
}

// Per-row LN (wave32 per row) -> fp32 + bf16-padded copy; optionally also
// convert an un-normalized source to bf16-padded (for K/V path).
__global__ __launch_bounds__(256)
void k_ln_conv(const float* __restrict__ src, const float* __restrict__ g,
               const float* __restrict__ bta, float* __restrict__ out_f,
               bf16_t* __restrict__ out_b, const float* __restrict__ conv_src,
               bf16_t* __restrict__ conv_dst) {
  int row = blockIdx.x * 8 + (threadIdx.x >> 5);
  int l = threadIdx.x & 31;
  const float* x = src + row * D_;
  float x0 = x[l];
  float x1 = (l + 32 < D_) ? x[l + 32] : 0.f;
  float mean = wred_sum(x0 + x1) * (1.0f / D_);
  float d0 = x0 - mean;
  float d1 = (l + 32 < D_) ? (x1 - mean) : 0.f;
  float var = wred_sum(d0 * d0 + d1 * d1) * (1.0f / D_);
  float inv = rsqrtf(var + 1e-8f);
  float y0 = d0 * inv * g[l] + bta[l];
  float y1 = (l + 32 < D_) ? (d1 * inv * g[l + 32] + bta[l + 32]) : 0.f;
  out_f[row * D_ + l] = y0;
  if (l + 32 < D_) out_f[row * D_ + l + 32] = y1;
  out_b[row * DP_ + l] = (bf16_t)y0;
  out_b[row * DP_ + l + 32] = (l + 32 < D_) ? (bf16_t)y1 : (bf16_t)0.f;
  if (conv_src != nullptr) {
    conv_dst[row * DP_ + l] = (bf16_t)conv_src[row * D_ + l];
    conv_dst[row * DP_ + l + 32] =
        (l + 32 < D_) ? (bf16_t)conv_src[row * D_ + l + 32] : (bf16_t)0.f;
  }
}

// Pack weights (fp32 row-major [Dout<=50][Din<=50]) into WMMA B-fragment layout:
// frag[nt][ks][lane][16] with B[k][n] = W[n][k], zero-padded to 64x64.
__global__ __launch_bounds__(32)
void k_pack(const float* __restrict__ in_w, const float* __restrict__ out_w,
            const float* __restrict__ f1_w, const float* __restrict__ f2_w,
            bf16_t* __restrict__ WqkvP, bf16_t* __restrict__ WoutP,
            bf16_t* __restrict__ F1P, bf16_t* __restrict__ F2P) {
  int id = blockIdx.x;
  const float* W; bf16_t* dst; int nt, ks2;
  if (id < 480) {                       // blk,e,mat,nt,ks
    int t = id;
    ks2 = t & 1; t >>= 1; nt = t & 3; t >>= 2;
    int mat = t % 3; t /= 3;
    int e = t % E_; t /= E_;
    int blk = t;
    W   = in_w + (((blk * E_ + e) * 3 + mat) * D_) * D_;
    dst = WqkvP + ((((blk * E_ + e) * 3 + mat) * 4 + nt) * 2 + ks2) * 512;
  } else if (id < 640) {
    int t = id - 480;
    ks2 = t & 1; t >>= 1; nt = t & 3; t >>= 2;
    int e = t % E_; t /= E_;
    int blk = t;
    W   = out_w + (blk * E_ + e) * (D_ * D_);
    dst = WoutP + (((blk * E_ + e) * 4 + nt) * 2 + ks2) * 512;
  } else if (id < 656) {
    int t = id - 640;
    ks2 = t & 1; t >>= 1; nt = t & 3; t >>= 2;
    int blk = t;
    W   = f1_w + blk * (D_ * D_);
    dst = F1P + ((blk * 4 + nt) * 2 + ks2) * 512;
  } else {
    int t = id - 656;
    ks2 = t & 1; t >>= 1; nt = t & 3; t >>= 2;
    int blk = t;
    W   = f2_w + blk * (D_ * D_);
    dst = F2P + ((blk * 4 + nt) * 2 + ks2) * 512;
  }
  int l = threadIdx.x;
  int n = nt * 16 + (l & 15);
  int kb = ks2 * 32 + (l >> 4) * 16;
#pragma unroll
  for (int j = 0; j < 16; ++j) {
    int k = kb + j;
    float v = (n < D_ && k < D_) ? W[n * D_ + k] : 0.f;
    dst[l * 16 + j] = (bf16_t)v;
  }
}

// QKV projections: q = LN(seqs)@Wq^T+bq, k/v = seqs@W^T+b. One wave per 16x16 tile.
__global__ __launch_bounds__(32)
void k_qkv(const bf16_t* __restrict__ Qb, const bf16_t* __restrict__ Sb,
           const bf16_t* __restrict__ WqkvP_blk, const float* __restrict__ in_b_blk,
           bf16_t* __restrict__ qb, bf16_t* __restrict__ kb, bf16_t* __restrict__ vb) {
  int mt  = blockIdx.x;                 // 0..1599 (BL/16)
  int nt  = blockIdx.y & 3;
  int mat = blockIdx.y >> 2;            // 0=q,1=k,2=v
  int e   = blockIdx.z;
  const bf16_t* A = ((mat == 0) ? Qb : Sb) + mt * 16 * DP_;
  const bf16_t* W = WqkvP_blk + (((e * 3 + mat) * 4 + nt) * 2) * 512;
  v8f c = {};
#pragma unroll
  for (int ks2 = 0; ks2 < 2; ++ks2) {
    v16bf a = load_a_frag(A + ks2 * 32, DP_);
    v16bf b = load_b_frag_packed(W + ks2 * 512);
    c = wmma_bf16(a, b, c);
  }
  int l = threadIdx.x;
  int n = nt * 16 + (l & 15);
  int mo = (l >> 4) * 8;
  if (n < D_) {
    float bias = in_b_blk[e * (3 * D_) + mat * D_ + n];
    bf16_t* dst = (mat == 0) ? qb : ((mat == 1) ? kb : vb);
#pragma unroll
    for (int r = 0; r < 8; ++r) {
      int row = mt * 16 + r + mo;
      dst[(size_t)(e * BL_ + row) * DP_ + n] = (bf16_t)(c[r] + bias);
    }
  }
}

// Fused attention for one (b,e,h,q-tile): async K/V/Q panel loads to LDS ->
// scores WMMA -> causal softmax -> ctx WMMA.
__global__ __launch_bounds__(32)
void k_attn(const bf16_t* __restrict__ qb, const bf16_t* __restrict__ kb,
            const bf16_t* __restrict__ vb, bf16_t* __restrict__ ctxb) {
  __shared__ bf16_t ks[LP_ * 32];       // keys   [key][dh pad 32]
  __shared__ bf16_t vs[LP_ * 32];       // values [key][dh pad 32]
  __shared__ bf16_t qs[16 * 32];        // query panel
  __shared__ float  sc[16 * LP_];       // score panel fp32
  __shared__ bf16_t at[16 * LP_];       // attn panel bf16
  int qt = blockIdx.x;                  // 0..12
  int h  = blockIdx.y % H_;
  int b  = blockIdx.y / H_;
  int e  = blockIdx.z;
  int l  = threadIdx.x;
  size_t rowbase = (size_t)(e * B_ + b) * L_;

  // --- zero ONLY the padding region via DS (disjoint from async targets) ---
  uint32_t* ks32 = (uint32_t*)ks;
  uint32_t* vs32 = (uint32_t*)vs;
  uint32_t* qs32 = (uint32_t*)qs;
  for (int i = l; i < LP_ * 11; i += 32) {   // dwords 5..15 (dh 10..31) all rows
    int row = i / 11, c = 5 + i % 11;
    ks32[row * 16 + c] = 0u;
    vs32[row * 16 + c] = 0u;
  }
  for (int i = l; i < (LP_ - L_) * 5; i += 32) {  // rows >= L_: dwords 0..4
    int row = L_ + i / 5, c = i % 5;
    ks32[row * 16 + c] = 0u;
    vs32[row * 16 + c] = 0u;
  }
  for (int i = l; i < 16 * 11; i += 32) {
    int r = i / 11, c = 5 + i % 11;
    qs32[r * 16 + c] = 0u;
  }
  // --- async copy the valid dh columns (5 dwords / row), memory -> LDS ---
  const bf16_t* ksrc = kb + rowbase * DP_ + h * DH_;
  const bf16_t* vsrc = vb + rowbase * DP_ + h * DH_;
  for (int i = l; i < L_ * 5; i += 32) {
    int row = i / 5, c = i % 5;
    async_g2l_b32(ksrc + (size_t)row * DP_ + c * 2, &ks[row * 32 + c * 2]);
    async_g2l_b32(vsrc + (size_t)row * DP_ + c * 2, &vs[row * 32 + c * 2]);
  }
  for (int i = l; i < 16 * 5; i += 32) {
    int r = i / 5, c = i % 5;
    int q = qt * 16 + r;
    if (q < L_)
      async_g2l_b32(qb + (rowbase + q) * DP_ + h * DH_ + c * 2, &qs[r * 32 + c * 2]);
    else
      qs32[r * 16 + c] = 0u;
  }
  async_wait_all();

  // scores (only lower-triangular key tiles)
  v16bf aq = load_a_frag(qs, 32);
  int n = l & 15, mo = (l >> 4) * 8, kb2 = (l >> 4) * 16;
  for (int jt = 0; jt <= qt; ++jt) {
    v16bf bk;
#pragma unroll
    for (int j = 0; j < 16; ++j) bk[j] = ks[(jt * 16 + n) * 32 + kb2 + j]; // k^T frag
    v8f c = {};
    c = wmma_bf16(aq, bk, c);
#pragma unroll
    for (int r = 0; r < 8; ++r) {
      int qrow = qt * 16 + r + mo;
      int key  = jt * 16 + n;
      float s = c[r] * RSQRT_DH;
      if (key > qrow) s = -1e30f;
      sc[(r + mo) * LP_ + key] = s;
    }
  }

  // causal softmax per row (wave-cooperative)
  for (int r = 0; r < 16; ++r) {
    int q = qt * 16 + r;
    if (q >= L_) {
      for (int c = l; c < LP_; c += 32) at[r * LP_ + c] = (bf16_t)0.f;
      continue;
    }
    float mx = -1e30f;
    for (int c = l; c <= q; c += 32) mx = fmaxf(mx, sc[r * LP_ + c]);
    mx = wred_max(mx);
    float sum = 0.f;
    for (int c = l; c <= q; c += 32) {
      float ev = __expf(sc[r * LP_ + c] - mx);
      sc[r * LP_ + c] = ev;
      sum += ev;
    }
    sum = wred_sum(sum);
    float inv = 1.0f / sum;
    for (int c = l; c < LP_; c += 32)
      at[r * LP_ + c] = (c <= q) ? (bf16_t)(sc[r * LP_ + c] * inv) : (bf16_t)0.f;
  }

  // ctx = attn @ v
  v8f c = {};
  for (int kt = 0; kt <= qt; ++kt) {
    v16bf a = load_a_frag(at + kt * 32, LP_);
    v16bf b = load_b_frag_rows(vs + kt * 32 * 32, 32);
    c = wmma_bf16(a, b, c);
  }
  if (n < DH_) {
#pragma unroll
    for (int r = 0; r < 8; ++r) {
      int q = qt * 16 + r + mo;
      if (q < L_) ctxb[(rowbase + q) * DP_ + h * DH_ + n] = (bf16_t)c[r];
    }
  }
}

// o = ctx@Wout^T+bo per expert, gate-weighted sum over experts, + residual Q.
__global__ __launch_bounds__(32)
void k_outproj(const bf16_t* __restrict__ ctxb, const bf16_t* __restrict__ WoutP_blk,
               const float* __restrict__ out_b_blk, const float* __restrict__ gate,
               const float* __restrict__ Qln, float* __restrict__ seqs) {
  int rt = blockIdx.x;                  // 0..12
  int nt = blockIdx.y;                  // 0..3
  int bi = blockIdx.z;                  // batch
  int l = threadIdx.x;
  int n = nt * 16 + (l & 15);
  int mo = (l >> 4) * 8;
  float acc[8] = {0.f, 0.f, 0.f, 0.f, 0.f, 0.f, 0.f, 0.f};
  for (int e = 0; e < E_; ++e) {
    if (e + 1 < E_) {   // hide next expert's operand latency behind this WMMA chain
      __builtin_prefetch(ctxb + ((size_t)((e + 1) * B_ + bi) * L_ + rt * 16) * DP_, 0, 0);
      __builtin_prefetch(WoutP_blk + (((e + 1) * 4 + nt) * 2) * 512, 0, 0);
    }
    float g = gate[bi * E_ + e];
    const bf16_t* A = ctxb + ((size_t)(e * B_ + bi) * L_ + rt * 16) * DP_;
    const bf16_t* W = WoutP_blk + ((e * 4 + nt) * 2) * 512;
    v8f c = {};
#pragma unroll
    for (int ks2 = 0; ks2 < 2; ++ks2) {
      v16bf a = load_a_frag(A + ks2 * 32, DP_);
      v16bf b = load_b_frag_packed(W + ks2 * 512);
      c = wmma_bf16(a, b, c);
    }
    float bias = (n < D_) ? out_b_blk[e * D_ + n] : 0.f;
#pragma unroll
    for (int r = 0; r < 8; ++r) acc[r] += g * (c[r] + bias);
  }
  if (n < D_) {
#pragma unroll
    for (int r = 0; r < 8; ++r) {
      int row = rt * 16 + r + mo;
      if (row < L_) {
        int gr = bi * L_ + row;
        seqs[gr * D_ + n] = Qln[gr * D_ + n] + acc[r];
      }
    }
  }
}

// Hb = relu(S2b@f1^T + b1), zero-padded to 64 cols (K-padding for ffn2).
__global__ __launch_bounds__(32)
void k_ffn1(const bf16_t* __restrict__ S2b, const bf16_t* __restrict__ F1P_blk,
            const float* __restrict__ f1_b_blk, bf16_t* __restrict__ Hb) {
  int mt = blockIdx.x, nt = blockIdx.y;
  const bf16_t* A = S2b + mt * 16 * DP_;
  const bf16_t* W = F1P_blk + (nt * 2) * 512;
  v8f c = {};
#pragma unroll
  for (int ks2 = 0; ks2 < 2; ++ks2) {
    v16bf a = load_a_frag(A + ks2 * 32, DP_);
    v16bf b = load_b_frag_packed(W + ks2 * 512);
    c = wmma_bf16(a, b, c);
  }
  int l = threadIdx.x;
  int n = nt * 16 + (l & 15);
  int mo = (l >> 4) * 8;
#pragma unroll
  for (int r = 0; r < 8; ++r) {
    int row = mt * 16 + r + mo;
    float v = (n < D_) ? fmaxf(c[r] + f1_b_blk[n], 0.f) : 0.f;
    Hb[(size_t)row * DP_ + n] = (bf16_t)v;
  }
}

// seqs = S2 + Hb@f2^T + b2
__global__ __launch_bounds__(32)
void k_ffn2(const bf16_t* __restrict__ Hb, const bf16_t* __restrict__ F2P_blk,
            const float* __restrict__ f2_b_blk, const float* __restrict__ S2,
            float* __restrict__ seqs) {
  int mt = blockIdx.x, nt = blockIdx.y;
  const bf16_t* A = Hb + mt * 16 * DP_;
  const bf16_t* W = F2P_blk + (nt * 2) * 512;
  v8f c = {};
#pragma unroll
  for (int ks2 = 0; ks2 < 2; ++ks2) {
    v16bf a = load_a_frag(A + ks2 * 32, DP_);
    v16bf b = load_b_frag_packed(W + ks2 * 512);
    c = wmma_bf16(a, b, c);
  }
  int l = threadIdx.x;
  int n = nt * 16 + (l & 15);
  int mo = (l >> 4) * 8;
  if (n < D_) {
#pragma unroll
    for (int r = 0; r < 8; ++r) {
      int row = mt * 16 + r + mo;
      seqs[row * D_ + n] = S2[row * D_ + n] + c[r] + f2_b_blk[n];
    }
  }
}

// feats = LNf(seqs); pos/neg logits via dot with gathered item embeddings.
__global__ __launch_bounds__(256)
void k_final(const float* __restrict__ seqs, const float* __restrict__ lnf_g,
             const float* __restrict__ lnf_b, const float* __restrict__ item_emb,
             const int* __restrict__ pos_seqs, const int* __restrict__ neg_seqs,
             float* __restrict__ out) {
  int row = blockIdx.x * 8 + (threadIdx.x >> 5);
  int l = threadIdx.x & 31;
  const float* x = seqs + row * D_;
  float x0 = x[l];
  float x1 = (l + 32 < D_) ? x[l + 32] : 0.f;
  float mean = wred_sum(x0 + x1) * (1.0f / D_);
  float d0 = x0 - mean;
  float d1 = (l + 32 < D_) ? (x1 - mean) : 0.f;
  float var = wred_sum(d0 * d0 + d1 * d1) * (1.0f / D_);
  float inv = rsqrtf(var + 1e-8f);
  float f0 = d0 * inv * lnf_g[l] + lnf_b[l];
  float f1 = (l + 32 < D_) ? (d1 * inv * lnf_g[l + 32] + lnf_b[l + 32]) : 0.f;
  const float* pe = item_emb + (size_t)pos_seqs[row] * D_;
  const float* ne = item_emb + (size_t)neg_seqs[row] * D_;
  float dp = f0 * pe[l] + ((l + 32 < D_) ? f1 * pe[l + 32] : 0.f);
  float dn = f0 * ne[l] + ((l + 32 < D_) ? f1 * ne[l + 32] : 0.f);
  dp = wred_sum(dp);
  dn = wred_sum(dn);
  if (l == 0) { out[row] = dp; out[BL_ + row] = dn; }
}

// ---------------- launch ----------------
extern "C" void kernel_launch(void* const* d_in, const int* in_sizes, int n_in,
                              void* d_out, int out_size, void* d_ws, size_t ws_size,
                              hipStream_t stream) {
  (void)in_sizes; (void)n_in; (void)out_size; (void)ws_size;
  const float* item_emb = (const float*)d_in[0];
  const float* pos_emb  = (const float*)d_in[1];
  const float* rW1      = (const float*)d_in[2];
  const float* rb1      = (const float*)d_in[3];
  const float* rW2      = (const float*)d_in[4];
  const float* rb2      = (const float*)d_in[5];
  const float* ln1_g    = (const float*)d_in[6];
  const float* ln1_b    = (const float*)d_in[7];
  const float* in_w     = (const float*)d_in[8];
  const float* in_b     = (const float*)d_in[9];
  const float* out_w    = (const float*)d_in[10];
  const float* out_b    = (const float*)d_in[11];
  const float* ln2_g    = (const float*)d_in[12];
  const float* ln2_b    = (const float*)d_in[13];
  const float* f1_w     = (const float*)d_in[14];
  const float* f1_b     = (const float*)d_in[15];
  const float* f2_w     = (const float*)d_in[16];
  const float* f2_b     = (const float*)d_in[17];
  const float* lnf_g    = (const float*)d_in[18];
  const float* lnf_b    = (const float*)d_in[19];
  const int*   log_seqs = (const int*)d_in[21];
  const int*   pos_seqs = (const int*)d_in[22];
  const int*   neg_seqs = (const int*)d_in[23];
  float* out = (float*)d_out;

  // workspace carve (~170 MB total)
  char* p = (char*)d_ws;
  auto alloc = [&](size_t bytes) {
    void* r = (void*)p;
    p += (bytes + 255) & ~(size_t)255;
    return r;
  };
  float*  seqs   = (float*)alloc((size_t)BL_ * D_ * 4);
  float*  Qln    = (float*)alloc((size_t)BL_ * D_ * 4);
  float*  S2     = (float*)alloc((size_t)BL_ * D_ * 4);
  float*  pooled = (float*)alloc((size_t)B_ * D_ * 4);
  float*  gate   = (float*)alloc((size_t)B_ * E_ * 4);
  bf16_t* Qb     = (bf16_t*)alloc((size_t)BL_ * DP_ * 2);
  bf16_t* Sb     = (bf16_t*)alloc((size_t)BL_ * DP_ * 2);
  bf16_t* S2b    = (bf16_t*)alloc((size_t)BL_ * DP_ * 2);
  bf16_t* Hb     = (bf16_t*)alloc((size_t)BL_ * DP_ * 2);
  bf16_t* qb     = (bf16_t*)alloc((size_t)E_ * BL_ * DP_ * 2);
  bf16_t* kb     = (bf16_t*)alloc((size_t)E_ * BL_ * DP_ * 2);
  bf16_t* vb     = (bf16_t*)alloc((size_t)E_ * BL_ * DP_ * 2);
  bf16_t* ctxb   = (bf16_t*)alloc(((size_t)E_ * BL_ + 16) * DP_ * 2); // +slack rows
  bf16_t* WqkvP  = (bf16_t*)alloc((size_t)NB_ * E_ * 3 * 8 * 512 * 2);
  bf16_t* WoutP  = (bf16_t*)alloc((size_t)NB_ * E_ * 8 * 512 * 2);
  bf16_t* F1P    = (bf16_t*)alloc((size_t)NB_ * 8 * 512 * 2);
  bf16_t* F2P    = (bf16_t*)alloc((size_t)NB_ * 8 * 512 * 2);

  k_embed<<<dim3(L_, B_), 64, 0, stream>>>(item_emb, pos_emb, log_seqs, seqs);
  k_pool<<<B_, 64, 0, stream>>>(item_emb, log_seqs, pooled);
  k_router<<<B_, 128, 0, stream>>>(pooled, rW1, rb1, rW2, rb2, gate);
  k_pack<<<672, 32, 0, stream>>>(in_w, out_w, f1_w, f2_w, WqkvP, WoutP, F1P, F2P);

  for (int blk = 0; blk < NB_; ++blk) {
    // Q = LN1(seqs) -> Qln/Qb ; Sb = bf16(seqs) (K/V source, un-normalized)
    k_ln_conv<<<BL_ / 8, 256, 0, stream>>>(seqs, ln1_g + blk * D_, ln1_b + blk * D_,
                                           Qln, Qb, seqs, Sb);
    k_qkv<<<dim3(BL_ / 16, 12, E_), 32, 0, stream>>>(
        Qb, Sb, WqkvP + (size_t)blk * E_ * 3 * 8 * 512, in_b + (size_t)blk * E_ * 3 * D_,
        qb, kb, vb);
    k_attn<<<dim3(13, B_ * H_, E_), 32, 0, stream>>>(qb, kb, vb, ctxb);
    k_outproj<<<dim3(13, 4, B_), 32, 0, stream>>>(
        ctxb, WoutP + (size_t)blk * E_ * 8 * 512, out_b + (size_t)blk * E_ * D_,
        gate, Qln, seqs);
    // seqs <- LN2(seqs) = S2 (also residual base), + bf16 copy
    k_ln_conv<<<BL_ / 8, 256, 0, stream>>>(seqs, ln2_g + blk * D_, ln2_b + blk * D_,
                                           S2, S2b, nullptr, nullptr);
    k_ffn1<<<dim3(BL_ / 16, 4), 32, 0, stream>>>(S2b, F1P + (size_t)blk * 8 * 512,
                                                 f1_b + blk * D_, Hb);
    k_ffn2<<<dim3(BL_ / 16, 4), 32, 0, stream>>>(Hb, F2P + (size_t)blk * 8 * 512,
                                                 f2_b + blk * D_, S2, seqs);
  }

  k_final<<<BL_ / 8, 256, 0, stream>>>(seqs, lnf_g, lnf_b, item_emb,
                                       pos_seqs, neg_seqs, out);
}
